// SimpleGSphereNetModel_37220186587498
// MI455X (gfx1250) — matrix operand order repaired and checked by reading, not compiled, not run on hardware
//
#include <hip/hip_runtime.h>
#include <hip/hip_bf16.h>
#include <cstdint>
#include <cstddef>

// Problem constants (from reference)
#define BN_NODES 8192     // B*N
#define DMODEL   512
#define RBF_DIM  64
#define ANG_DIM  32
#define EF_DIM   96       // RBF_DIM + ANG_DIM
#define EF_PAD   128      // K padded to a multiple of 64 for the GEMM pipeline
#define N_LAYERS 4

typedef _Float16 half_t;
typedef _Float16 v16h __attribute__((ext_vector_type(16)));
typedef _Float16 v8h  __attribute__((ext_vector_type(8)));
typedef float    v8f  __attribute__((ext_vector_type(8)));
typedef int      v4i  __attribute__((ext_vector_type(4)));

#if __has_builtin(__builtin_amdgcn_global_load_async_to_lds_b128) && \
    __has_builtin(__builtin_amdgcn_s_wait_asynccnt)
#define HAVE_ASYNC_LDS 1
#else
#define HAVE_ASYNC_LDS 0
#endif

typedef __attribute__((address_space(1))) v4i* gv4i_p;
typedef __attribute__((address_space(3))) v4i* lv4i_p;

// 16-byte global -> LDS copy. Async (ASYNCcnt-tracked, no VGPR round trip)
// when the CDNA5 builtins exist; otherwise a register-staged ds_store.
__device__ __forceinline__ void cp16_g2l(const half_t* g, half_t* l) {
#if HAVE_ASYNC_LDS
  __builtin_amdgcn_global_load_async_to_lds_b128((gv4i_p)(void*)g,
                                                 (lv4i_p)(void*)l, 0, 0);
#else
  *(v8h*)l = *(const v8h*)g;
#endif
}

template <int N>
__device__ __forceinline__ void cp_wait() {
#if HAVE_ASYNC_LDS
  __builtin_amdgcn_s_wait_asynccnt(N);
#endif
}

// ---------------------------------------------------------------- utilities

__global__ __launch_bounds__(256) void f32_to_f16_kernel(
    const float* __restrict__ s, half_t* __restrict__ d, int n) {
  for (int i = blockIdx.x * blockDim.x + threadIdx.x; i < n;
       i += gridDim.x * blockDim.x)
    d[i] = (half_t)s[i];
}

// ep_w [L][96][512] f32 -> [L][128][512] f16, rows 96..127 zero
__global__ __launch_bounds__(256) void ep_pad_f16_kernel(
    const float* __restrict__ s, half_t* __restrict__ d, int n) {
  for (int i = blockIdx.x * blockDim.x + threadIdx.x; i < n;
       i += gridDim.x * blockDim.x) {
    int l   = i / (EF_PAD * DMODEL);
    int rem = i - l * (EF_PAD * DMODEL);
    int k   = rem / DMODEL;
    int c   = rem - k * DMODEL;
    d[i] = (k < EF_DIM)
               ? (half_t)s[((size_t)l * EF_DIM + k) * DMODEL + c]
               : (half_t)0.0f;
  }
}

__global__ __launch_bounds__(256) void embed_gather_kernel(
    const int* __restrict__ tok, const float* __restrict__ emb,
    float* __restrict__ x) {
  int i = blockIdx.x * blockDim.x + threadIdx.x;   // exact grid: BN*D threads
  int node = i >> 9;            // / DMODEL
  int f    = i & (DMODEL - 1);  // % DMODEL
  x[i] = emb[(size_t)tok[node] * DMODEL + f];
}

__global__ __launch_bounds__(256) void mask_kernel(
    const int* __restrict__ tok, float* __restrict__ m) {
  int i = blockIdx.x * blockDim.x + threadIdx.x;
  if (i < BN_NODES) m[i] = (tok[i] == 0) ? 1.0f : 0.0f;
}

// ------------------------------------------------- edge features (RBF part)

__global__ __launch_bounds__(256) void edge_rbf_kernel(
    const float* __restrict__ coord, const int* __restrict__ eidx,
    const float* __restrict__ centers, float* __restrict__ ef, int E) {
  int total = E * EF_DIM;
  for (int i = blockIdx.x * blockDim.x + threadIdx.x; i < total;
       i += gridDim.x * blockDim.x) {
    int e = i / EF_DIM;
    int f = i - e * EF_DIM;
    float v = 0.0f;                       // angle slots zero-initialized here
    if (f < RBF_DIM) {
      int r = eidx[e], c = eidx[E + e];
      float dx = coord[3 * r + 0] - coord[3 * c + 0];
      float dy = coord[3 * r + 1] - coord[3 * c + 1];
      float dz = coord[3 * r + 2] - coord[3 * c + 2];
      float dist = sqrtf(dx * dx + dy * dy + dz * dz);
      float t = dist - centers[f];
      v = expf(-10.0f * t * t);
    }
    ef[i] = v;
  }
}

// ------------------------------------------- triplet angle MLP + scatter-add

__global__ __launch_bounds__(256) void triplet_kernel(
    const float* __restrict__ coord, const int* __restrict__ eidx,
    const int* __restrict__ te1, const int* __restrict__ te2,
    const float* __restrict__ aw1, const float* __restrict__ ab1,
    const float* __restrict__ aw2, const float* __restrict__ ab2,
    float* __restrict__ ef, int E, int T) {
  long total = (long)T * ANG_DIM;   // one thread per (triplet, out-dim)
  for (long i = blockIdx.x * (long)blockDim.x + threadIdx.x; i < total;
       i += (long)gridDim.x * blockDim.x) {
    int t  = (int)(i >> 5);
    int jd = (int)(i & 31);
    int ea = te1[t], eb = te2[t];
    int nj = eidx[ea];        // center j  (row of e1)
    int ni = eidx[E + ea];    // neighbor i
    int nk = eidx[E + eb];    // neighbor k
    float jx = coord[3 * nj + 0], jy = coord[3 * nj + 1], jz = coord[3 * nj + 2];
    float ix = coord[3 * ni + 0] - jx, iy = coord[3 * ni + 1] - jy, iz = coord[3 * ni + 2] - jz;
    float kx = coord[3 * nk + 0] - jx, ky = coord[3 * nk + 1] - jy, kz = coord[3 * nk + 2] - jz;
    float dot = ix * kx + iy * ky + iz * kz;
    float n1 = sqrtf(ix * ix + iy * iy + iz * iz);
    float n2 = sqrtf(kx * kx + ky * ky + kz * kz);
    float denom = fmaxf(n1 * n2, 1e-8f);
    float cosv = fminf(fmaxf(dot / denom, -1.0f), 1.0f);
    float acc = ab2[jd];
    #pragma unroll
    for (int m = 0; m < ANG_DIM; ++m) {
      float h = fmaxf(cosv * aw1[m] + ab1[m], 0.0f);
      acc += h * aw2[m * ANG_DIM + jd];
    }
    atomicAdd(&ef[(size_t)ea * EF_DIM + RBF_DIM + jd], acc);
    atomicAdd(&ef[(size_t)eb * EF_DIM + RBF_DIM + jd], acc);
  }
}

// --------------------- CSR offsets via binary search (row array is sorted)

__global__ __launch_bounds__(256) void node_csr_kernel(
    const int* __restrict__ rows, int* __restrict__ off,
    float* __restrict__ degf, int E) {
  int n = blockIdx.x * blockDim.x + threadIdx.x;
  if (n > BN_NODES) return;
  int lo = 0, hi = E;
  while (lo < hi) { int mid = (lo + hi) >> 1; if (rows[mid] < n) lo = mid + 1; else hi = mid; }
  off[n] = lo;
  if (n < BN_NODES) {
    int lo2 = lo, hi2 = E;
    while (lo2 < hi2) { int mid = (lo2 + hi2) >> 1; if (rows[mid] < n + 1) lo2 = mid + 1; else hi2 = mid; }
    degf[n] = (float)(lo2 - lo);
  }
}

// ----- per-node aggregation of 96-dim edge features (done ONCE, all layers)
// Output is K-padded to 128 halves (zeros in 96..127).

__global__ void node_ef_kernel(const float* __restrict__ ef,
                               const int* __restrict__ off,
                               half_t* __restrict__ nef) {
  int n = blockIdx.x;
  int f = threadIdx.x;          // 128 threads
  float s = 0.0f;
  if (f < EF_DIM) {
    int e0 = off[n], e1 = off[n + 1];
    for (int e = e0; e < e1; ++e) s += ef[(size_t)e * EF_DIM + f];
  }
  nef[(size_t)n * EF_PAD + f] = (half_t)s;
}

// ----------------------------------------------------------- WMMA GEMM
// Block: 256 threads (8 wave32), tile M=64 x N=128, K-step 64 (2x 16x16x32).
// LDS double buffer: A 64x64 (pad->72), B 64x128 (pad->136) per stage.
// Stage k+1 via async-to-LDS while WMMA consumes stage k from LDS.
// All 10 fragments (2 A + 8 B) preloaded, then 8 back-to-back WMMAs.
#define A_LDS 72    // 64 + 8 halves pad (row stride 144B, 16B aligned)
#define B_LDS 136   // 128 + 8 halves pad (row stride 272B, 16B aligned)
#define CPS 6       // async copies issued per thread per stage (2 A + 4 B)

// MODE 0: out16 = f16(A@W + rowscale[row]*bias[col])       (edge projection)
// MODE 1: out16 = f16(relu(A@W + bias[col]))               (u1)
// MODE 2: out32 += A@W + bias[col]                         (u2, residual)
template <int MODE>
__global__ __launch_bounds__(256) void gemm_wmma_kernel(
    const half_t* __restrict__ A, const half_t* __restrict__ W,
    const float* __restrict__ bias, const float* __restrict__ rowscale,
    half_t* __restrict__ out16, float* __restrict__ out32, int M, int K,
    int Nn) {
  __shared__ half_t As[2][64][A_LDS];
  __shared__ half_t Bs[2][64][B_LDS];

  const int t     = threadIdx.x;
  const int lane  = t & 31;
  const int wave  = t >> 5;
  const int waveM = wave & 3;       // 4 waves along M
  const int waveN = wave >> 2;      // 2 waves along N
  const int mblk  = blockIdx.y * 64;
  const int nblk  = blockIdx.x * 128;
  const int m0    = mblk + waveM * 16;
  const int n0    = nblk + waveN * 64;

  // staging coordinates (256 threads move A:8KB + B:16KB per stage)
  const int srow = t >> 2;          // 0..63 (shared by A and B staging)
  const int acol = (t & 3) * 16;    // A: 64 rows x 64 halves, 2 cp/thread
  const int bcol = (t & 3) * 32;    // B: 64 rows x 128 halves, 4 cp/thread

  auto stage = [&](int buf, int k0) {
    const half_t* ag = &A[(size_t)(mblk + srow) * K + k0 + acol];
    cp16_g2l(ag + 0, &As[buf][srow][acol + 0]);
    cp16_g2l(ag + 8, &As[buf][srow][acol + 8]);
    const half_t* bg = &W[(size_t)(k0 + srow) * Nn + nblk + bcol];
    cp16_g2l(bg + 0,  &Bs[buf][srow][bcol + 0]);
    cp16_g2l(bg + 8,  &Bs[buf][srow][bcol + 8]);
    cp16_g2l(bg + 16, &Bs[buf][srow][bcol + 16]);
    cp16_g2l(bg + 24, &Bs[buf][srow][bcol + 24]);
  };

  v8f acc[4] = {v8f{}, v8f{}, v8f{}, v8f{}};
  const int nk = K >> 6;            // K-step 64

  const int arowc = waveM * 16 + (lane & 15);
  const int acolc = (lane >> 4) * 8;
  const int bncol = waveN * 64;

  stage(0, 0);
  for (int ks = 0; ks < nk; ++ks) {
    const int cur = ks & 1;
    if (ks + 1 < nk) {            // issue next stage, then drain current one
      stage(cur ^ 1, (ks + 1) * 64);
      cp_wait<CPS>();             // async loads complete in order
    } else {
      cp_wait<0>();
    }
    __syncthreads();

    // Preload ALL fragments, then burst the 8 WMMAs back-to-back.
    v16h afrag[2];
    v16h bfrag[2][4];
    #pragma unroll
    for (int s = 0; s < 2; ++s) {
      // A fragment (ISA 7.12.2): lane m=lane&15, hi=lane>>4;
      // elems 0..7 = K[hi*8..+7], elems 8..15 = K[16+hi*8..+7]
      const half_t* ap = &As[cur][arowc][s * 32 + acolc];
      v8h alo = *(const v8h*)ap;
      v8h ahi = *(const v8h*)(ap + 16);
      afrag[s] = __builtin_shufflevector(alo, ahi, 0, 1, 2, 3, 4, 5, 6, 7, 8,
                                         9, 10, 11, 12, 13, 14, 15);
      #pragma unroll
      for (int nt = 0; nt < 4; ++nt) {
        // B fragment: lane holds col N=lane&15; elems 0..7 = K[hi*8..+7], 8..15 = K[16+hi*8..+7]
        const half_t* bp = &Bs[cur][s * 32 + (lane >> 4) * 8][bncol + nt * 16 + (lane & 15)];
        #pragma unroll
        for (int i = 0; i < 8; ++i) {
          bfrag[s][nt][i] = bp[i * B_LDS];
          bfrag[s][nt][8 + i] = bp[(16 + i) * B_LDS];
        }
      }
    }
    #pragma unroll
    for (int s = 0; s < 2; ++s)
      #pragma unroll
      for (int nt = 0; nt < 4; ++nt)
        acc[nt] = __builtin_amdgcn_wmma_f32_16x16x32_f16(
            false, afrag[s], false, bfrag[s][nt], (short)0, acc[nt], false,
            false);
    __syncthreads();              // protect buffer `cur` before it is re-staged
  }

  const int hi   = lane >> 4;
  const int mrow = m0 + (hi ? 8 : 0);
  float rs[8];
  if (MODE == 0) {
    #pragma unroll
    for (int r = 0; r < 8; ++r) rs[r] = rowscale[mrow + r];
  }
  #pragma unroll
  for (int nt = 0; nt < 4; ++nt) {
    int col = n0 + nt * 16 + (lane & 15);
    float bcolv = bias[col];
    #pragma unroll
    for (int r = 0; r < 8; ++r) {
      size_t idx = (size_t)(mrow + r) * Nn + col;
      float v = acc[nt][r];
      if (MODE == 0) {
        out16[idx] = (half_t)(v + rs[r] * bcolv);
      } else if (MODE == 1) {
        v += bcolv;
        out16[idx] = (half_t)(v > 0.0f ? v : 0.0f);
      } else {
        out32[idx] += v + bcolv;
      }
    }
  }
}

// ------------------------------------------------------------------- driver

extern "C" void kernel_launch(void* const* d_in, const int* in_sizes, int n_in,
                              void* d_out, int out_size, void* d_ws,
                              size_t ws_size, hipStream_t stream) {
  (void)n_in; (void)out_size; (void)ws_size;
  const int*   tok   = (const int*)d_in[0];
  const float* coord = (const float*)d_in[1];
  const int*   eidx  = (const int*)d_in[2];
  const int*   te1   = (const int*)d_in[3];
  const int*   te2   = (const int*)d_in[4];
  const float* emb   = (const float*)d_in[5];
  const float* cent  = (const float*)d_in[6];
  const float* aw1   = (const float*)d_in[7];
  const float* ab1   = (const float*)d_in[8];
  const float* aw2   = (const float*)d_in[9];
  const float* ab2   = (const float*)d_in[10];
  const float* ep_w  = (const float*)d_in[11];
  const float* ep_b  = (const float*)d_in[12];
  const float* u1_w  = (const float*)d_in[13];
  const float* u1_b  = (const float*)d_in[14];
  const float* u2_w  = (const float*)d_in[15];
  const float* u2_b  = (const float*)d_in[16];
  const int E = in_sizes[2] / 2;
  const int T = in_sizes[3];

  float* x       = (float*)d_out;                  // [BN, D], updated in place
  float* maskout = x + (size_t)BN_NODES * DMODEL;  // [BN] padding mask

  // workspace carve-out (256B aligned so every 16B vector access is aligned)
  uintptr_t p = (uintptr_t)d_ws;
  auto take = [&](size_t bytes) -> void* {
    uintptr_t q = (p + 255) & ~(uintptr_t)255;
    p = q + bytes;
    return (void*)q;
  };
  half_t* wep  = (half_t*)take((size_t)N_LAYERS * EF_PAD * DMODEL * sizeof(half_t));
  half_t* wu1  = (half_t*)take((size_t)N_LAYERS * DMODEL * DMODEL * sizeof(half_t));
  half_t* wu2  = (half_t*)take((size_t)N_LAYERS * DMODEL * DMODEL * sizeof(half_t));
  int*    off  = (int*)take((size_t)(BN_NODES + 1) * sizeof(int));
  float*  degf = (float*)take((size_t)BN_NODES * sizeof(float));
  half_t* nef  = (half_t*)take((size_t)BN_NODES * EF_PAD * sizeof(half_t));
  half_t* agg  = (half_t*)take((size_t)BN_NODES * DMODEL * sizeof(half_t));
  half_t* hbuf = (half_t*)take((size_t)BN_NODES * DMODEL * sizeof(half_t));
  float*  ef   = (float*)take((size_t)(E > 0 ? E : 1) * EF_DIM * sizeof(float));

  // weights -> f16 (L2-resident, reused every layer); ep_w zero-padded K->128
  {
    int n = N_LAYERS * EF_PAD * DMODEL;
    ep_pad_f16_kernel<<<(n + 255) / 256, 256, 0, stream>>>(ep_w, wep, n);
    n = N_LAYERS * DMODEL * DMODEL;
    f32_to_f16_kernel<<<(n + 255) / 256, 256, 0, stream>>>(u1_w, wu1, n);
    f32_to_f16_kernel<<<(n + 255) / 256, 256, 0, stream>>>(u2_w, wu2, n);
  }

  embed_gather_kernel<<<(BN_NODES * DMODEL) / 256, 256, 0, stream>>>(tok, emb, x);
  mask_kernel<<<(BN_NODES + 255) / 256, 256, 0, stream>>>(tok, maskout);

  if (E > 0) {
    int total = E * EF_DIM;
    edge_rbf_kernel<<<(total + 255) / 256, 256, 0, stream>>>(coord, eidx, cent, ef, E);
    if (T > 0) {
      long total32 = (long)T * ANG_DIM;
      int blocks = (int)((total32 + 255) / 256);
      triplet_kernel<<<blocks, 256, 0, stream>>>(coord, eidx, te1, te2, aw1,
                                                 ab1, aw2, ab2, ef, E, T);
    }
  }
  node_csr_kernel<<<(BN_NODES + 1 + 255) / 256, 256, 0, stream>>>(eidx, off, degf, E);
  node_ef_kernel<<<BN_NODES, EF_PAD, 0, stream>>>(ef, off, nef);

  // layer loop: 3 WMMA GEMMs per layer
  dim3 g(DMODEL / 128, BN_NODES / 64);   // (4, 128) blocks, 256 threads each
  for (int l = 0; l < N_LAYERS; ++l) {
    gemm_wmma_kernel<0><<<g, 256, 0, stream>>>(
        nef, wep + (size_t)l * EF_PAD * DMODEL, ep_b + l * DMODEL, degf, agg,
        nullptr, BN_NODES, EF_PAD, DMODEL);
    gemm_wmma_kernel<1><<<g, 256, 0, stream>>>(
        agg, wu1 + (size_t)l * DMODEL * DMODEL, u1_b + l * DMODEL, nullptr,
        hbuf, nullptr, BN_NODES, DMODEL, DMODEL);
    gemm_wmma_kernel<2><<<g, 256, 0, stream>>>(
        hbuf, wu2 + (size_t)l * DMODEL * DMODEL, u2_b + l * DMODEL, nullptr,
        nullptr, x, BN_NODES, DMODEL, DMODEL);
  }
}
